// Custom_loss_28621662060738
// MI455X (gfx1250) — matrix operand order repaired and checked by reading, not compiled, not run on hardware
//
#include <hip/hip_runtime.h>
#include <hip/hip_bf16.h>

// ---------------------------------------------------------------------------
// predict (21,512,512) f32, feature (64,512,512) f32, R=5, scalar f32 out.
// d2(p,q) = |p|^2 + |q|^2 - 2 p.q ; cross-dots via banded GEMM: one
// 16x64 x 64x32 f16 WMMA GEMM per (16-pixel strip, vertical offset m)
// yields all 10 horizontal offsets. Feature map (32MB f16) is L2-resident.
// ---------------------------------------------------------------------------
#define H 512
#define W 512
#define C 64
#define NCLS 21
#define NPIX (H * W)

typedef __attribute__((ext_vector_type(16))) _Float16 v16h;
typedef __attribute__((ext_vector_type(8)))  float    v8f;

union V16 { v16h v; uint4 q[2]; };

struct BFrag { V16 lo0, hi0, lo1, hi1; };   // 2 column tiles x 2 K-halves

__device__ __forceinline__ int iclamp(int v, int lo, int hi) {
    return v < lo ? lo : (v > hi ? hi : v);
}

// Same-wave LDS store->load edge: DS ops are hardware in-order within a wave;
// wait only the DS counter (NOT loadcnt, so global prefetches stay in flight).
__device__ __forceinline__ void lds_store_fence() {
    asm volatile("s_wait_dscnt 0x0" ::: "memory");
}
// Load->store edge: hardware keeps same-wave DS ops ordered; only stop the
// compiler from reordering. Zero instructions.
__device__ __forceinline__ void compiler_fence() {
    asm volatile("" ::: "memory");
}

// ---------------------------------------------------------------------------
__global__ void k_zero(float* out) {
    if (threadIdx.x == 0 && blockIdx.x == 0) out[0] = 0.0f;
}

// meta[p] = (nrm, cls-as-float-bits). argmax fills .y, convert fills .x
__global__ void __launch_bounds__(256)
k_argmax(const float* __restrict__ predict, float* __restrict__ meta) {
    int p = blockIdx.x * blockDim.x + threadIdx.x;
    if (p >= NPIX) return;
    float best = predict[p];
    int   bi   = 0;
    #pragma unroll
    for (int c = 1; c < NCLS; ++c) {
        float v = predict[c * NPIX + p];
        if (v > best) { best = v; bi = c; }     // first-max, like jnp.argmax
    }
    meta[2 * p + 1] = __int_as_float(bi);
}

__global__ void __launch_bounds__(256)
k_convert(const float* __restrict__ feat, _Float16* __restrict__ fh,
          float* __restrict__ meta) {
    int p = blockIdx.x * blockDim.x + threadIdx.x;
    if (p >= NPIX) return;
    float acc = 0.0f;
    #pragma unroll 8
    for (int c = 0; c < C; ++c) {
        float v = feat[c * NPIX + p];           // coalesced per channel
        _Float16 h = (_Float16)v;
        fh[(size_t)p * C + c] = h;
        float hv = (float)h;
        acc += hv * hv;                          // norm from f16-rounded vals
    }
    meta[2 * p] = acc;
}

// ---------------------------------------------------------------------------
// B 32x16 f16 fragment as loaded here (k order matches A): lanes0-15 col N=lid K=0..15, lanes16-31
// K=16..31 -> 16 contiguous halves per lane (two b128 loads each).
// ---------------------------------------------------------------------------
__device__ __forceinline__ void loadB(const _Float16* __restrict__ fh,
                                      int qyc, int x0, int lid, int half,
                                      BFrag& b) {
    const int qx0 = iclamp(x0 - 5 + lid, 0, W - 1);
    const int qx1 = iclamp(x0 + 11 + lid, 0, W - 1);
    const _Float16* bp0 = fh + ((size_t)qyc * W + qx0) * C + half * 16;
    const _Float16* bp1 = fh + ((size_t)qyc * W + qx1) * C + half * 16;
    b.lo0.q[0] = *(const uint4*)(bp0);       b.lo0.q[1] = *(const uint4*)(bp0 + 8);
    b.hi0.q[0] = *(const uint4*)(bp0 + 32);  b.hi0.q[1] = *(const uint4*)(bp0 + 40);
    b.lo1.q[0] = *(const uint4*)(bp1);       b.lo1.q[1] = *(const uint4*)(bp1 + 8);
    b.hi1.q[0] = *(const uint4*)(bp1 + 32);  b.hi1.q[1] = *(const uint4*)(bp1 + 40);
}

// ---------------------------------------------------------------------------
__global__ void __launch_bounds__(32)
k_main(const _Float16* __restrict__ fh, const float* __restrict__ meta,
       float* __restrict__ out) {
    __shared__ float Dt[16][32];

    const int lane = threadIdx.x;
    const int half = lane >> 4;               // 0 or 1
    const int lid  = lane & 15;               // pixel-in-strip / tile column
    const int y    = blockIdx.y;
    const int x0   = blockIdx.x * 16;

    // ---- A fragments (pixels x0..x0+15 of row y), fixed for the block ----
    // A 16x32 f16 as loaded here (k order matches B): lanes0-15 K=0..15, lanes16-31 K=16..31
    const size_t pOff = ((size_t)y * W + x0 + lid) * C;
    const int c0 = half * 16;
    V16 a_lo, a_hi;
    a_lo.q[0] = *(const uint4*)(fh + pOff + c0);
    a_lo.q[1] = *(const uint4*)(fh + pOff + c0 + 8);
    a_hi.q[0] = *(const uint4*)(fh + pOff + c0 + 32);
    a_hi.q[1] = *(const uint4*)(fh + pOff + c0 + 40);

    const float2 mp  = ((const float2*)meta)[y * W + x0 + lid];
    const float nrmp = mp.x;
    const int   clsp = __float_as_int(mp.y);

    float ssum = 0.0f, csum = 0.0f;

    static const int MS[10] = {-5, -4, -3, -2, -1, 1, 2, 3, 4, 5};

    BFrag cur, nxt;
    loadB(fh, iclamp(y + MS[0], 0, H - 1), x0, lid, half, cur);

    #pragma unroll
    for (int mi = 0; mi < 10; ++mi) {
        const int  m     = MS[mi];
        const int  qy    = y + m;
        const int  qyc   = iclamp(qy, 0, H - 1);
        const bool rowOK = (qy >= 0) && (qy < H);

        // Epilogue meta loads FIRST: LOADcnt retires in order, so waiting on
        // these 5 small loads in the epilogue leaves the 8 B-fragment
        // prefetches (issued after) still in flight across the epilogue.
        float2 mq[5];
        #pragma unroll
        for (int k = 0; k < 5; ++k) {
            const int n  = half ? (k + 1) : (k - 5);
            const int qi = qyc * W + iclamp(x0 + lid + n, 0, W - 1);
            mq[k] = ((const float2*)meta)[qi];
        }

        // Prefetch next iteration's B fragments.
        if (mi < 9)
            loadB(fh, iclamp(y + MS[mi + 1], 0, H - 1), x0, lid, half, nxt);

        // ---- 4 WMMAs: two 16x16 D tiles, K=64 in two x32 steps.
        // Operands (cur) completed during the previous epilogue: no stall.
        v8f acc0 = {}, acc1 = {};
        acc0 = __builtin_amdgcn_wmma_f32_16x16x32_f16(
                   false, a_lo.v, false, cur.lo0.v, (short)0, acc0, false, false);
        acc0 = __builtin_amdgcn_wmma_f32_16x16x32_f16(
                   false, a_hi.v, false, cur.hi0.v, (short)0, acc0, false, false);
        acc1 = __builtin_amdgcn_wmma_f32_16x16x32_f16(
                   false, a_lo.v, false, cur.lo1.v, (short)0, acc1, false, false);
        acc1 = __builtin_amdgcn_wmma_f32_16x16x32_f16(
                   false, a_hi.v, false, cur.hi1.v, (short)0, acc1, false, false);

        // D layout: lane l, VGPR r -> D[r + 8*(l>=16)][l&15]
        #pragma unroll
        for (int r = 0; r < 8; ++r) {
            Dt[r + 8 * half][lid]      = acc0[r];
            Dt[r + 8 * half][lid + 16] = acc1[r];
        }
        lds_store_fence();     // s_wait_dscnt 0 only; B prefetch stays in flight

        // ---- epilogue: lanes0-15 do n=-5..-1, lanes16-31 n=+1..+5 ----
        #pragma unroll
        for (int k = 0; k < 5; ++k) {
            const int   n_neg = k - 5, n_pos = k + 1;
            const int   n     = half ? n_pos : n_neg;
            const int   j     = lid + n + 5;            // 0..25
            const float dotv  = Dt[lid][j];
            const int   qx2   = x0 + lid + n;
            const bool  ok    = rowOK && ((unsigned)qx2 < (unsigned)W);
            const bool  msk   = ok && (__float_as_int(mq[k].y) != clsp);
            const float d2    = fmaxf(nrmp + mq[k].x - 2.0f * dotv, 0.0f);
            const float d     = __builtin_amdgcn_sqrtf(d2);   // raw v_sqrt_f32
            // dist(m,n): constant-folded per unrolled (m,k), cndmask on half
            const float dn_   = __builtin_sqrtf((float)(m * m + n_neg * n_neg));
            const float dp_   = __builtin_sqrtf((float)(m * m + n_pos * n_pos));
            const float dist  = half ? dp_ : dn_;
            const float wgt   = __expf(-d * dist);
            ssum += msk ? wgt : 0.0f;
            csum += msk ? 1.0f : 0.0f;
        }
        compiler_fence();      // DS in-order in HW; just stop IR reordering

        cur = nxt;             // SSA-renamed away by the full unroll
    }

    // Combine the two lanes handling the same pixel, reduce, one atomic.
    float stot = ssum + __shfl_xor(ssum, 16, 32);
    float ctot = csum + __shfl_xor(csum, 16, 32);
    float pp = (lane < 16) ? (stot / fmaxf(ctot, 1.0f)) : 0.0f;
    #pragma unroll
    for (int off = 16; off >= 1; off >>= 1)
        pp += __shfl_xor(pp, off, 32);
    if (lane == 0)
        atomicAdd(out, pp * (1.0f / (float)NPIX));
}

// ---------------------------------------------------------------------------
// Workspace: fh (32MB f16) | meta (2MB float2: nrm + cls bits)
// ---------------------------------------------------------------------------
extern "C" void kernel_launch(void* const* d_in, const int* in_sizes, int n_in,
                              void* d_out, int out_size, void* d_ws, size_t ws_size,
                              hipStream_t stream) {
    const float* predict = (const float*)d_in[0];   // (1,21,512,512) f32
    // d_in[1] = pseudo_label (unused by the reference loss)
    const float* feature = (const float*)d_in[2];   // (1,64,512,512) f32
    float* out = (float*)d_out;

    char* ws = (char*)d_ws;
    _Float16* fh   = (_Float16*)ws;                                      // 32 MB
    float*    meta = (float*)(ws + (size_t)NPIX * C * sizeof(_Float16)); //  2 MB

    k_zero<<<1, 1, 0, stream>>>(out);
    k_argmax<<<NPIX / 256, 256, 0, stream>>>(predict, meta);
    k_convert<<<NPIX / 256, 256, 0, stream>>>(feature, fh, meta);

    dim3 grid(W / 16, H);   // 32 x 512 strips, one wave32 each
    k_main<<<grid, 32, 0, stream>>>(fh, meta, out);
}
